// Head_15590731284780
// MI455X (gfx1250) — matrix operand order repaired and checked by reading, not compiled, not run on hardware
//
#include <hip/hip_runtime.h>
#include <hip/hip_bf16.h>
#include <math.h>

typedef __attribute__((ext_vector_type(16))) __bf16 v16bf;
typedef __attribute__((ext_vector_type(8)))  __bf16 v8bf;
typedef __attribute__((ext_vector_type(8)))  float  v8f;

#define Bn 4
#define Tn 4096
#define Cn 1024
#define Hn 64
#define NSPLIT 4

#if __has_builtin(__builtin_amdgcn_sched_barrier)
#define SCHED_FENCE() __builtin_amdgcn_sched_barrier(0)
#else
#define SCHED_FENCE() asm volatile("" ::: "memory")
#endif

static __device__ __forceinline__ v8f wmma_bf16(v16bf a, v16bf b, v8f c) {
    // D = A(16x32 bf16) * B(32x16 bf16) + C(16x16 f32)
    return __builtin_amdgcn_wmma_f32_16x16x32_bf16(
        /*neg_a=*/false, a, /*neg_b=*/false, b,
        /*c_mod=*/(short)0, c, /*reuse_a=*/false, /*reuse_b=*/false);
}

// ---------------------------------------------------------------------------
// Kernel 0: convert & transpose weights.
//   Wq/Wk/Wv: [C,H] f32  ->  Wt: [3][H][C] bf16  (1/sqrt(C)=1/32 folded into Wq)
// ---------------------------------------------------------------------------
__global__ void __launch_bounds__(256) wt_kernel(const float* __restrict__ Wq,
                                                 const float* __restrict__ Wk,
                                                 const float* __restrict__ Wv,
                                                 __bf16* __restrict__ Wt) {
    int idx = blockIdx.x * 256 + threadIdx.x;          // < 3*C*H
    const int CH = Cn * Hn;
    int mat = idx / CH;
    int rem = idx - mat * CH;                          // rem = c*H + h (coalesced read)
    int c = rem >> 6;
    int h = rem & 63;
    const float* W = (mat == 0) ? Wq : ((mat == 1) ? Wk : Wv);
    float v = W[rem] * ((mat == 0) ? 0.03125f : 1.0f);
    Wt[(size_t)mat * CH + (size_t)h * Cn + c] = (__bf16)v;
}

// ---------------------------------------------------------------------------
// Kernel 1: fused QKV projection with WMMA.
//   Each wave: 16 rows x 192 cols.  All 24 B-fragment b128 loads are issued,
//   then a sched_barrier pins them ahead of the 12-WMMA drain so a single L2
//   latency is amortized over the whole K-step.
// ---------------------------------------------------------------------------
__global__ void __launch_bounds__(256) qkv_kernel(const float*  __restrict__ x,
                                                  const __bf16* __restrict__ Wt,
                                                  __bf16* __restrict__ Qb,
                                                  __bf16* __restrict__ Kb,
                                                  __bf16* __restrict__ Vt) {
    const int lane = threadIdx.x & 31;
    const int wave = threadIdx.x >> 5;
    const int l16  = lane & 15;
    const int grp  = lane >> 4;              // 0: lanes 0-15, 1: lanes 16-31
    const int tile = blockIdx.x * 8 + wave;  // 0..B*T/16-1
    const int row0 = tile * 16;

    const float* xrow = x + (size_t)(row0 + l16) * Cn;
    const int aoff = grp ? 16 : 0;
    const int boff = grp ? 16 : 0;

    // One lane-dependent B base; per-tile offset is the uniform j*16*Cn
    // (folds into the 24-bit instruction offset field).
    const __bf16* bbase = Wt + (size_t)l16 * Cn + boff;

    v8f acc[12];
#pragma unroll
    for (int j = 0; j < 12; ++j) acc[j] = (v8f){0,0,0,0,0,0,0,0};

    for (int k0 = 0; k0 < Cn; k0 += 32) {
        // A fragment (16x32 bf16): two 8-element chunks, f32 -> bf16
        const float* pa0 = xrow + k0 + aoff;
        const float* pa1 = xrow + k0 + aoff + 8;
        v8bf lo8, hi8;
#pragma unroll
        for (int i = 0; i < 8; ++i) {
            lo8[i] = (__bf16)pa0[i];
            hi8[i] = (__bf16)pa1[i];
        }
        v16bf a = __builtin_shufflevector(lo8, hi8,
                     0,1,2,3,4,5,6,7,8,9,10,11,12,13,14,15);

        // Preload ALL B fragments ...
        v16bf barr[12];
#pragma unroll
        for (int j = 0; j < 12; ++j)
            barr[j] = *(const v16bf*)(bbase + (size_t)j * (16 * Cn) + k0);
        // ... fence so the loads stay ahead of the WMMA drain (keeps 24 loads
        // in flight; waits become partial s_wait_loadcnt instead of 12x full
        // latency stalls) ...
        SCHED_FENCE();
#pragma unroll
        for (int j = 0; j < 12; ++j)
            acc[j] = wmma_bf16(a, barr[j], acc[j]);
    }

    // ---- stores ----
#pragma unroll
    for (int j = 0; j < 8; ++j) {
        __bf16* dst = (j < 4) ? Qb : Kb;
        int hcol = (j & 3) * 16 + l16;
#pragma unroll
        for (int r = 0; r < 8; ++r) {
            int row = row0 + r + 8 * grp;
            dst[(size_t)row * Hn + hcol] = (__bf16)acc[j][r];
        }
    }
    {
        int bidx = row0 >> 12;           // /T
        int t0   = row0 & (Tn - 1);
#pragma unroll
        for (int j = 8; j < 12; ++j) {
            int h = (j - 8) * 16 + l16;
            v8bf pk;
#pragma unroll
            for (int r = 0; r < 8; ++r) pk[r] = (__bf16)acc[j][r];
            *(v8bf*)(Vt + (size_t)bidx * Hn * Tn + (size_t)h * Tn + t0 + 8 * grp) = pk;
        }
    }
}

// ---------------------------------------------------------------------------
// Kernel 2: split-K causal flash attention.
//   Wave = (query tile of 16 rows, split s of 4).  Key tiles of 64, interleaved
//   j0 = 64*(s + 4*i) for causal load balance.  Writes unnormalized partials
//   (O, m, l); combine kernel merges.  -INFINITY masking + finite floor on
//   m_new makes empty/fully-masked tiles contribute exact zeros.
// ---------------------------------------------------------------------------
__global__ void __launch_bounds__(256) attn_kernel(const __bf16* __restrict__ Qb,
                                                   const __bf16* __restrict__ Kb,
                                                   const __bf16* __restrict__ Vt,
                                                   float* __restrict__ Opart,
                                                   float* __restrict__ Mpart,
                                                   float* __restrict__ Lpart) {
    __shared__ __align__(16) __bf16 psh[8][16 * 64];   // 2KB per wave

    const int lane  = threadIdx.x & 31;
    const int wave  = threadIdx.x >> 5;
    const int l16   = lane & 15;
    const int grp   = lane >> 4;
    const int gid   = blockIdx.x * 8 + wave;           // 0..4095
    const int qtile = gid >> 2;                        // 0..1023
    const int split = gid & 3;
    const int bidx  = qtile >> 8;                      // qtile*16 / T
    const int tq0   = (qtile << 4) & (Tn - 1);

    const int aoff = grp ? 16 : 0;
    const int boff = grp ? 16 : 0;

    // Resident Q A-fragments (16 rows x 64) = 2 fragments
    const __bf16* qrow = Qb + (size_t)(bidx * Tn + tq0 + l16) * Hn;
    v16bf qf[2];
#pragma unroll
    for (int kk = 0; kk < 2; ++kk) {
        v8bf lo = *(const v8bf*)(qrow + kk * 32 + aoff);
        v8bf hi = *(const v8bf*)(qrow + kk * 32 + aoff + 8);
        qf[kk] = __builtin_shufflevector(lo, hi,
                     0,1,2,3,4,5,6,7,8,9,10,11,12,13,14,15);
    }

    const __bf16* Kbase = Kb + (size_t)bidx * Tn * Hn;
    const __bf16* Vbase = Vt + (size_t)bidx * Hn * Tn;

    float m_i[8], l_i[8];
#pragma unroll
    for (int r = 0; r < 8; ++r) { m_i[r] = -INFINITY; l_i[r] = 0.0f; }
    v8f oacc[4];
#pragma unroll
    for (int n = 0; n < 4; ++n) oacc[n] = (v8f){0,0,0,0,0,0,0,0};

    const int jend = tq0 + 16;                          // keys needed: 0..tq0+15
    for (int j0 = split * 64; j0 < jend; j0 += NSPLIT * 64) {
        // ---- preload all 8 K fragments, fence, then 8 S-WMMAs ----
        v16bf kfrag[4][2];
#pragma unroll
        for (int t = 0; t < 4; ++t) {
            const __bf16* kp = Kbase + (size_t)(j0 + t * 16 + l16) * Hn + boff;
            kfrag[t][0] = *(const v16bf*)kp;
            kfrag[t][1] = *(const v16bf*)(kp + 32);
        }
        SCHED_FENCE();
        v8f st[4];
#pragma unroll
        for (int t = 0; t < 4; ++t) {
            v8f z = (v8f){0,0,0,0,0,0,0,0};
            z = wmma_bf16(qf[0], kfrag[t][0], z);
            st[t] = wmma_bf16(qf[1], kfrag[t][1], z);
        }

        // ---- causal mask + tile row max ----
        float tmax[8];
#pragma unroll
        for (int r = 0; r < 8; ++r) tmax[r] = -INFINITY;
#pragma unroll
        for (int t = 0; t < 4; ++t) {
            int col = j0 + t * 16 + l16;
#pragma unroll
            for (int r = 0; r < 8; ++r) {
                int rowg = tq0 + r + 8 * grp;
                float s = st[t][r];
                if (col > rowg) s = -INFINITY;
                st[t][r] = s;
                tmax[r] = fmaxf(tmax[r], s);
            }
        }
#pragma unroll
        for (int r = 0; r < 8; ++r) {
#pragma unroll
            for (int m = 1; m < 16; m <<= 1)
                tmax[r] = fmaxf(tmax[r], __shfl_xor(tmax[r], m, 32));
        }

        // ---- online softmax update (finite floor keeps exp well-defined) ----
        float corr[8], rsum[8];
#pragma unroll
        for (int r = 0; r < 8; ++r) {
            float mn = fmaxf(fmaxf(m_i[r], tmax[r]), -1.0e38f);
            corr[r] = __expf(m_i[r] - mn);     // exp(-inf)=0 on first real tile
            m_i[r] = mn;
            rsum[r] = 0.0f;
        }
#pragma unroll
        for (int t = 0; t < 4; ++t) {
#pragma unroll
            for (int r = 0; r < 8; ++r) {
                float p = __expf(st[t][r] - m_i[r]);   // masked: exp(-inf)=0
                st[t][r] = p;
                rsum[r] += p;
            }
        }
#pragma unroll
        for (int r = 0; r < 8; ++r) {
#pragma unroll
            for (int m = 1; m < 16; m <<= 1)
                rsum[r] += __shfl_xor(rsum[r], m, 32);
            l_i[r] = l_i[r] * corr[r] + rsum[r];
        }
#pragma unroll
        for (int n = 0; n < 4; ++n)
#pragma unroll
            for (int r = 0; r < 8; ++r)
                oacc[n][r] *= corr[r];

        // ---- P: C/D f32 layout -> bf16 A-fragment layout via per-wave LDS ----
#pragma unroll
        for (int t = 0; t < 4; ++t) {
            int col = t * 16 + l16;
#pragma unroll
            for (int r = 0; r < 8; ++r)
                psh[wave][(r + 8 * grp) * 64 + col] = (__bf16)st[t][r];
        }

        // ---- V^T B-fragments: issue while LDS round-trip is in flight ----
        v16bf vf[4][2];
#pragma unroll
        for (int n = 0; n < 4; ++n)
#pragma unroll
            for (int kk = 0; kk < 2; ++kk)
                vf[n][kk] = *(const v16bf*)(Vbase + (size_t)(n * 16 + l16) * Tn
                                            + j0 + kk * 32 + boff);

        asm volatile("s_wait_dscnt 0" ::: "memory");   // lockstep wave32
        const __bf16* pb = &psh[wave][l16 * 64 + aoff];
        v8bf p0 = *(const v8bf*)pb;
        v8bf p1 = *(const v8bf*)(pb + 8);
        v8bf p2 = *(const v8bf*)(pb + 32);
        v8bf p3 = *(const v8bf*)(pb + 40);
        v16bf pf0 = __builtin_shufflevector(p0, p1,
                        0,1,2,3,4,5,6,7,8,9,10,11,12,13,14,15);
        v16bf pf1 = __builtin_shufflevector(p2, p3,
                        0,1,2,3,4,5,6,7,8,9,10,11,12,13,14,15);

        // ---- O += P . V ----
#pragma unroll
        for (int n = 0; n < 4; ++n) {
            oacc[n] = wmma_bf16(pf0, vf[n][0], oacc[n]);
            oacc[n] = wmma_bf16(pf1, vf[n][1], oacc[n]);
        }
    }

    // ---- write unnormalized partials ----
    const int ps = qtile * NSPLIT + split;
#pragma unroll
    for (int n = 0; n < 4; ++n) {
        int hcol = n * 16 + l16;
#pragma unroll
        for (int r = 0; r < 8; ++r)
            Opart[((size_t)ps * 16 + r + 8 * grp) * Hn + hcol] = oacc[n][r];
    }
    if (l16 == 0) {
#pragma unroll
        for (int r = 0; r < 8; ++r) {
            Mpart[(size_t)ps * 16 + r + 8 * grp] = m_i[r];
            Lpart[(size_t)ps * 16 + r + 8 * grp] = l_i[r];
        }
    }
}

// ---------------------------------------------------------------------------
// Kernel 3: combine split-K partials.  One thread per output element.
// ---------------------------------------------------------------------------
__global__ void __launch_bounds__(256) combine_kernel(const float* __restrict__ Opart,
                                                      const float* __restrict__ Mpart,
                                                      const float* __restrict__ Lpart,
                                                      float* __restrict__ out) {
    int idx = blockIdx.x * 256 + threadIdx.x;   // < B*T*H
    int row = idx >> 6;                         // global row over B*T
    int h   = idx & 63;
    int q   = row >> 4;
    int r   = row & 15;

    float ms[NSPLIT];
    float M = -INFINITY;
#pragma unroll
    for (int s = 0; s < NSPLIT; ++s) {
        ms[s] = Mpart[(size_t)(q * NSPLIT + s) * 16 + r];
        M = fmaxf(M, ms[s]);
    }
    float L = 0.0f, acc = 0.0f;
#pragma unroll
    for (int s = 0; s < NSPLIT; ++s) {
        float w = __expf(ms[s] - M);            // empty split: exp(-inf)=0
        L   += w * Lpart[(size_t)(q * NSPLIT + s) * 16 + r];
        acc += w * Opart[((size_t)(q * NSPLIT + s) * 16 + r) * Hn + h];
    }
    out[idx] = acc / L;
}

// ---------------------------------------------------------------------------
extern "C" void kernel_launch(void* const* d_in, const int* in_sizes, int n_in,
                              void* d_out, int out_size, void* d_ws, size_t ws_size,
                              hipStream_t stream) {
    const float* x  = (const float*)d_in[0];
    const float* Wq = (const float*)d_in[1];
    const float* Wk = (const float*)d_in[2];
    const float* Wv = (const float*)d_in[3];
    float* out = (float*)d_out;

    // Workspace layout (bytes):
    //   Wt    : 3*H*C*2          =   393216
    //   Qb    : B*T*H*2          =  2097152
    //   Kb    : B*T*H*2          =  2097152
    //   Vt    : B*H*T*2          =  2097152
    //   Opart : 1024*4*16*64*4   = 16777216
    //   Mpart : 1024*4*16*4      =   262144
    //   Lpart : 1024*4*16*4      =   262144   (total ~23 MiB, L2 resident)
    char* ws = (char*)d_ws;
    constexpr size_t offQ  = 393216;
    constexpr size_t offK  = offQ + (size_t)2 * Bn * Tn * Hn;
    constexpr size_t offV  = offK + (size_t)2 * Bn * Tn * Hn;
    constexpr size_t offO  = offV + (size_t)2 * Bn * Tn * Hn;
    constexpr size_t offM  = offO + (size_t)(Bn * Tn / 16) * NSPLIT * 16 * Hn * 4;
    constexpr size_t offL  = offM + (size_t)(Bn * Tn / 16) * NSPLIT * 16 * 4;

    __bf16* Wt = (__bf16*)(ws);
    __bf16* Qb = (__bf16*)(ws + offQ);
    __bf16* Kb = (__bf16*)(ws + offK);
    __bf16* Vt = (__bf16*)(ws + offV);
    float* Opart = (float*)(ws + offO);
    float* Mpart = (float*)(ws + offM);
    float* Lpart = (float*)(ws + offL);

    wt_kernel     <<<(3 * Cn * Hn) / 256, 256, 0, stream>>>(Wq, Wk, Wv, Wt);
    qkv_kernel    <<<(Bn * Tn) / (16 * 8), 256, 0, stream>>>(x, Wt, Qb, Kb, Vt);
    attn_kernel   <<<(Bn * Tn / 16) * NSPLIT / 8, 256, 0, stream>>>(Qb, Kb, Vt,
                                                                    Opart, Mpart, Lpart);
    combine_kernel<<<(Bn * Tn * Hn) / 256, 256, 0, stream>>>(Opart, Mpart, Lpart, out);
}
